// MyNet_16741782520461
// MI455X (gfx1250) — matrix operand order repaired and checked by reading, not compiled, not run on hardware
//
#include <hip/hip_runtime.h>
#include <hip/hip_bf16.h>

// ---------------------------------------------------------------------------
// MI455X / gfx1250, wave32, bf16 WMMA (v_wmma_f32_16x16x32_bf16) +
// async global->LDS copies (global_load_async_to_lds_b128 / s_wait_asynccnt).
//
// Net: FC1 (16384x784 @ 784x1024) -> reshape (B,32,32) -> LSTM(H=64) ->
//      LSTM(H=64) -> last step -> FC2 (64->10).
//
//   prep:       cast weights f32->bf16 into ws, combine biases.
//   fc1_gemm:   block tile M=128 x N=64 (8 waves x 16 rows). B-panel (64x32)
//               double-buffered in LDS, streamed with async-to-LDS loads,
//               overlapped with WMMAs; A prefetched in registers.
//   lstm_fused: each wave owns 16 batch rows for all 32 timesteps of BOTH
//               LSTM layers + FC2. c in VGPR accumulators, h round-trips
//               through per-wave LDS scratch (D->A transpose). All weights
//               staged once per block into dynamic LDS via async loads.
// Workspace use: ~35.3 MB.
// ---------------------------------------------------------------------------

typedef __attribute__((ext_vector_type(16))) __bf16 bf16x16;
typedef __attribute__((ext_vector_type(8)))  __bf16 bf16x8;
typedef __attribute__((ext_vector_type(8)))  float  f32x8;
typedef __attribute__((ext_vector_type(4)))  float  f32x4;

#define WMMA_BF16(A, Bm, C) \
    __builtin_amdgcn_wmma_f32_16x16x32_bf16(false, (A), false, (Bm), (short)0, (C), false, false)

// Async global->LDS copy, 16B per lane. Low 32 bits of a generic LDS pointer
// are the LDS byte offset (aperture base lives in the high half).
static __device__ __forceinline__ void async_copy_b128(void* lds_dst, const void* gsrc) {
    uint32_t l = (uint32_t)(uintptr_t)lds_dst;
    asm volatile("global_load_async_to_lds_b128 %0, %1, off"
                 :: "v"(l), "v"(gsrc) : "memory");
}
#define ASYNC_WAIT_0 asm volatile("s_wait_asynccnt 0x0" ::: "memory")
#define ASYNC_WAIT_1 asm volatile("s_wait_asynccnt 0x1" ::: "memory")

static __device__ __forceinline__ bf16x16 cat16(bf16x8 lo, bf16x8 hi) {
    bf16x16 r;
#pragma unroll
    for (int j = 0; j < 8; ++j) { r[j] = lo[j]; r[j + 8] = hi[j]; }
    return r;
}

// A-operand (16x32 bf16, MxK): lane<16 holds row=lane, K = bk..bk+7 and
// bk+16..bk+23 with bk=0; lane>=16: same with bk=8.  p = &row[k_base].
static __device__ __forceinline__ bf16x16 ldA16(const __bf16* p, int bk) {
    return cat16(*(const bf16x8*)(p + bk), *(const bf16x8*)(p + bk + 16));
}

// B-operand (32x16 bf16, KxN): lane holds col n=lane&15; k = bk..bk+7 and
// bk+16..bk+23 (bk=0 lane<16, 8 lane>=16). B[k][n] = W[n][k] row-major.
static __device__ __forceinline__ bf16x16 ldB16(const __bf16* p) {
    return cat16(*(const bf16x8*)p, *(const bf16x8*)(p + 16));
}

// A-operand built from fp32 source (FC1 reads x directly).
static __device__ __forceinline__ bf16x16 ldA_f32(const float* p, int bk) {
    f32x4 a0 = *(const f32x4*)(p + bk),      a1 = *(const f32x4*)(p + bk + 4);
    f32x4 a2 = *(const f32x4*)(p + bk + 16), a3 = *(const f32x4*)(p + bk + 20);
    bf16x16 r;
#pragma unroll
    for (int j = 0; j < 4; ++j) {
        r[j]      = (__bf16)a0[j];  r[j + 4]  = (__bf16)a1[j];
        r[j + 8]  = (__bf16)a2[j];  r[j + 12] = (__bf16)a3[j];
    }
    return r;
}

// Tail (16 valid K): upper half of A zeroed, so the matching B values are
// multiplied by 0 and may be garbage.
static __device__ __forceinline__ bf16x16 ldA_f32_tail(const float* p, int bk) {
    f32x4 a0 = *(const f32x4*)(p + bk), a1 = *(const f32x4*)(p + bk + 4);
    bf16x16 r;
#pragma unroll
    for (int j = 0; j < 4; ++j) {
        r[j]     = (__bf16)a0[j];   r[j + 4]  = (__bf16)a1[j];
        r[j + 8] = (__bf16)0.f;     r[j + 12] = (__bf16)0.f;
    }
    return r;
}

static __device__ __forceinline__ f32x8 bc8(float v) {
    f32x8 r;
#pragma unroll
    for (int j = 0; j < 8; ++j) r[j] = v;
    return r;
}

static __device__ __forceinline__ float sigmf(float x) {
    return 1.f / (1.f + __expf(-x));
}
static __device__ __forceinline__ float tanh_fast(float x) {
    float t = __expf(-2.f * fabsf(x));       // in (0,1], no overflow
    float r = (1.f - t) / (1.f + t);
    return copysignf(r, x);
}

// ---------------------------------------------------------------------------
// Kernel 1: cast/pack weights + combine biases.
// ---------------------------------------------------------------------------
__global__ __launch_bounds__(256) void prep_weights(
    const float* fc1_w, const float* wih0, const float* whh0,
    const float* wih1,  const float* whh1, const float* fc2_w,
    const float* bih0,  const float* bhh0, const float* bih1, const float* bhh1,
    __bf16* d_fc1w, __bf16* d_wih0, __bf16* d_whh0, __bf16* d_wih1,
    __bf16* d_whh1, __bf16* d_fc2p, float* d_b0, float* d_b1)
{
    int i = blockIdx.x * 256 + threadIdx.x;
    if (i < 802816) { d_fc1w[i] = (__bf16)fc1_w[i]; return; }
    i -= 802816;
    if (i < 8192)   { d_wih0[i] = (__bf16)wih0[i];  return; }
    i -= 8192;
    if (i < 16384)  { d_whh0[i] = (__bf16)whh0[i];  return; }
    i -= 16384;
    if (i < 16384)  { d_wih1[i] = (__bf16)wih1[i];  return; }
    i -= 16384;
    if (i < 16384)  { d_whh1[i] = (__bf16)whh1[i];  return; }
    i -= 16384;
    if (i < 1024) {                       // fc2 padded 16x64 (rows >=10 zero)
        int n = i >> 6, k = i & 63;
        d_fc2p[i] = (__bf16)((n < 10) ? fc2_w[n * 64 + k] : 0.f);
        return;
    }
    i -= 1024;
    if (i < 256) { d_b0[i] = bih0[i] + bhh0[i]; return; }
    i -= 256;
    if (i < 256) { d_b1[i] = bih1[i] + bhh1[i]; }
}

// ---------------------------------------------------------------------------
// Kernel 2: H1[16384,1024] = x @ fc1_w^T + fc1_b   (bf16 out)
// Block tile M=128 x N=64; B-panel shared via double-buffered LDS (async).
// ---------------------------------------------------------------------------
__global__ __launch_bounds__(256) void fc1_gemm(
    const float* __restrict__ x, const float* __restrict__ fc1_b,
    const __bf16* __restrict__ wbf, __bf16* __restrict__ h1)
{
    __shared__ __bf16 sB[2][64 * 32];    // 2 x 4KB B panels

    const int tid  = threadIdx.x;
    const int lane = tid & 31;
    const int wv   = tid >> 5;
    const int n0   = (blockIdx.x & 15) * 64;
    const int m0   = ((blockIdx.x >> 4) * 8 + wv) * 16;
    const int col  = lane & 15;
    const int bkA  = (lane >= 16) ? 8 : 0;
    const int bkB  = (lane >= 16) ? 8 : 0;

    // staging map: thread i -> B row nl = i>>2, k-chunk kk = (i&3)*8
    const int snl = tid >> 2;
    const int skk = (tid & 3) * 8;
    const __bf16* sgsrc = wbf + (size_t)(n0 + snl) * 784 + skk;
    __bf16* sdst = &sB[0][0] + snl * 32 + skk;

    async_copy_b128(sdst, sgsrc);        // stage k-step 0

    f32x8 acc[4];
#pragma unroll
    for (int tn = 0; tn < 4; ++tn) acc[tn] = bc8(fc1_b[n0 + tn * 16 + col]);

    const float* xrow = x + (size_t)(m0 + col) * 784;
    bf16x16 aNext = ldA_f32(xrow, bkA);

    for (int s = 0; s < 25; ++s) {       // K = 24*32 + 16-tail
        const int cur = s & 1;
        bf16x16 a = aNext;
        if (s < 24) {
            async_copy_b128(&sB[cur ^ 1][0] + snl * 32 + skk,
                            sgsrc + (s + 1) * 32);
            aNext = (s == 23) ? ldA_f32_tail(xrow + 768, bkA)
                              : ldA_f32(xrow + (s + 1) * 32, bkA);
            ASYNC_WAIT_1;                // stage s complete (s+1 in flight)
        } else {
            ASYNC_WAIT_0;                // final stage complete
        }
        __syncthreads();                 // all waves staged panel s
        const __bf16* bb = &sB[cur][0];
#pragma unroll
        for (int tn = 0; tn < 4; ++tn)
            acc[tn] = WMMA_BF16(a, ldB16(bb + (tn * 16 + col) * 32 + bkB), acc[tn]);
        __syncthreads();                 // panel consumed before overwrite
    }

    const int rbase = (lane >= 16) ? 8 : 0;
#pragma unroll
    for (int tn = 0; tn < 4; ++tn)
#pragma unroll
        for (int j = 0; j < 8; ++j)
            h1[(size_t)(m0 + rbase + j) * 1024 + n0 + tn * 16 + col] =
                (__bf16)acc[tn][j];
}

// ---------------------------------------------------------------------------
// Fused LSTM helpers
// ---------------------------------------------------------------------------
template <int NX>
static __device__ __forceinline__ void gates4(
    f32x8 out[4], const float* breg, const bf16x16* ax,
    const bf16x16& ah0, const bf16x16& ah1,
    const __bf16* wih, int ldih, const __bf16* whh,
    int g, int col, int bkB)
{
#pragma unroll
    for (int tn = 0; tn < 4; ++tn) {
        const int n = g * 64 + tn * 16 + col;
        f32x8 a = bc8(breg[tn]);
        a = WMMA_BF16(ax[0], ldB16(wih + n * ldih + bkB), a);
        if (NX == 2)
            a = WMMA_BF16(ax[1], ldB16(wih + n * ldih + 32 + bkB), a);
        a = WMMA_BF16(ah0, ldB16(whh + n * 64 + bkB), a);
        a = WMMA_BF16(ah1, ldB16(whh + n * 64 + 32 + bkB), a);
        out[tn] = a;
    }
}

// One LSTM cell step for 16 batch rows owned by this wave.
template <int NX>
static __device__ __forceinline__ void lstm_step(
    const bf16x16* ax, const __bf16* wih, int ldih, const __bf16* whh,
    const float* breg, f32x8 c[4], __bf16* hb, int lane)
{
    const int col   = lane & 15;
    const int bkA   = (lane >= 16) ? 8 : 0;
    const int bkB   = (lane >= 16) ? 8 : 0;
    const int rbase = (lane >= 16) ? 8 : 0;

    const bf16x16 ah0 = ldA16(hb + col * 64,      bkA);   // K 0..31
    const bf16x16 ah1 = ldA16(hb + col * 64 + 32, bkA);   // K 32..63

    f32x8 ig[4], t4[4];
    gates4<NX>(ig, breg + 0, ax, ah0, ah1, wih, ldih, whh, 0, col, bkB);   // i
#pragma unroll
    for (int tn = 0; tn < 4; ++tn)
#pragma unroll
        for (int j = 0; j < 8; ++j) ig[tn][j] = sigmf(ig[tn][j]);

    gates4<NX>(t4, breg + 4, ax, ah0, ah1, wih, ldih, whh, 1, col, bkB);   // f
#pragma unroll
    for (int tn = 0; tn < 4; ++tn)
#pragma unroll
        for (int j = 0; j < 8; ++j) c[tn][j] = sigmf(t4[tn][j]) * c[tn][j];

    gates4<NX>(t4, breg + 8, ax, ah0, ah1, wih, ldih, whh, 2, col, bkB);   // g
#pragma unroll
    for (int tn = 0; tn < 4; ++tn)
#pragma unroll
        for (int j = 0; j < 8; ++j) c[tn][j] += ig[tn][j] * tanh_fast(t4[tn][j]);

    gates4<NX>(t4, breg + 12, ax, ah0, ah1, wih, ldih, whh, 3, col, bkB);  // o
#pragma unroll
    for (int tn = 0; tn < 4; ++tn)
#pragma unroll
        for (int j = 0; j < 8; ++j) {
            float h = sigmf(t4[tn][j]) * tanh_fast(c[tn][j]);
            hb[(rbase + j) * 64 + tn * 16 + col] = (__bf16)h;  // D->A transpose
        }
}

// ---------------------------------------------------------------------------
// Kernel 3: fused 2-layer LSTM (T=32) + FC2.  8 waves/block, wave = 16 rows.
// Dynamic LDS: wih0(16KB) whh0(32KB) wih1(32KB) whh1(32KB) fc2(2KB) + 32KB h.
// ---------------------------------------------------------------------------
__global__ __launch_bounds__(256) void lstm_fused(
    const __bf16* __restrict__ h1, const __bf16* __restrict__ wblock,
    const float* __restrict__ bias0, const float* __restrict__ bias1,
    const float* __restrict__ fc2_b, float* __restrict__ out)
{
    extern __shared__ char smem[];
    __bf16* s_wih0 = (__bf16*)smem;          // 8192  elems
    __bf16* s_whh0 = s_wih0 + 8192;          // 16384 elems
    __bf16* s_wih1 = s_whh0 + 16384;         // 16384 elems
    __bf16* s_whh1 = s_wih1 + 16384;         // 16384 elems
    __bf16* s_fc2  = s_whh1 + 16384;         // 1024  elems
    __bf16* s_hb   = s_fc2  + 1024;          // 8 waves * 2048 elems

    {   // async stage of the contiguous 116736-byte weight block (7296 x 16B)
        const char* src = (const char*)wblock;
        char* dst = (char*)smem;
        for (int i = threadIdx.x; i < 7296; i += 256)
            async_copy_b128(dst + (size_t)i * 16, src + (size_t)i * 16);
        ASYNC_WAIT_0;
    }
    __syncthreads();

    const int lane = threadIdx.x & 31;
    const int wv   = threadIdx.x >> 5;
    const int m0   = (blockIdx.x * 8 + wv) * 16;
    const int col  = lane & 15;
    const int bkA  = (lane >= 16) ? 8 : 0;

    __bf16* hb0 = s_hb + wv * 2048;          // layer0 h (16x64)
    __bf16* hb1 = hb0 + 1024;                // layer1 h (16x64)
    for (int i = lane; i < 2048; i += 32) hb0[i] = (__bf16)0.f;

    float br0[16], br1[16];                  // per-lane gate biases
#pragma unroll
    for (int i = 0; i < 16; ++i) {
        const int n = (i >> 2) * 64 + (i & 3) * 16 + col;
        br0[i] = bias0[n];
        br1[i] = bias1[n];
    }

    f32x8 c0[4], c1[4];
#pragma unroll
    for (int tn = 0; tn < 4; ++tn) { c0[tn] = bc8(0.f); c1[tn] = bc8(0.f); }

    const __bf16* xbase = h1 + (size_t)(m0 + col) * 1024;

    for (int t = 0; t < 32; ++t) {
        bf16x16 ax[2];
        ax[0] = ldA16(xbase + t * 32, bkA);                       // x_t (K=32)
        lstm_step<1>(ax, s_wih0, 32, s_whh0, br0, c0, hb0, lane); // layer 0

        ax[0] = ldA16(hb0 + col * 64,      bkA);                  // h0 (K=64)
        ax[1] = ldA16(hb0 + col * 64 + 32, bkA);
        lstm_step<2>(ax, s_wih1, 64, s_whh1, br1, c1, hb1, lane); // layer 1
    }

    // FC2 on last hidden state: out[16x10] = h1_last @ fc2_w^T + fc2_b
    const int bkB = (lane >= 16) ? 8 : 0;
    const bf16x16 ah0 = ldA16(hb1 + col * 64,      bkA);
    const bf16x16 ah1 = ldA16(hb1 + col * 64 + 32, bkA);
    float bv = fc2_b[(col < 10) ? col : 0];
    bv = (col < 10) ? bv : 0.f;
    f32x8 acc = bc8(bv);
    acc = WMMA_BF16(ah0, ldB16(s_fc2 + col * 64 + bkB), acc);
    acc = WMMA_BF16(ah1, ldB16(s_fc2 + col * 64 + 32 + bkB), acc);

    const int rbase = (lane >= 16) ? 8 : 0;
    if (col < 10) {
#pragma unroll
        for (int j = 0; j < 8; ++j)
            out[(size_t)(m0 + rbase + j) * 10 + col] = acc[j];
    }
}

// ---------------------------------------------------------------------------
extern "C" void kernel_launch(void* const* d_in, const int* in_sizes, int n_in,
                              void* d_out, int out_size, void* d_ws, size_t ws_size,
                              hipStream_t stream)
{
    (void)in_sizes; (void)n_in; (void)out_size; (void)ws_size;

    const float* x     = (const float*)d_in[0];
    const float* fc1_w = (const float*)d_in[1];
    const float* fc1_b = (const float*)d_in[2];
    const float* wih0  = (const float*)d_in[3];
    const float* whh0  = (const float*)d_in[4];
    const float* bih0  = (const float*)d_in[5];
    const float* bhh0  = (const float*)d_in[6];
    const float* wih1  = (const float*)d_in[7];
    const float* whh1  = (const float*)d_in[8];
    const float* bih1  = (const float*)d_in[9];
    const float* bhh1  = (const float*)d_in[10];
    const float* fc2_w = (const float*)d_in[11];
    const float* fc2_b = (const float*)d_in[12];
    float* out = (float*)d_out;

    // ---- workspace layout (~35.3 MB) ----
    char* w = (char*)d_ws;
    size_t off = 0;
    auto al256 = [](size_t v) { return (v + 255) & ~(size_t)255; };

    __bf16* p_fc1w = (__bf16*)(w + off); off = al256(off + (size_t)(802816 + 64) * 2);
    // contiguous weight block (copied to LDS as one range):
    __bf16* p_wih0 = (__bf16*)(w + off); off += 8192  * 2;
    __bf16* p_whh0 = (__bf16*)(w + off); off += 16384 * 2;
    __bf16* p_wih1 = (__bf16*)(w + off); off += 16384 * 2;
    __bf16* p_whh1 = (__bf16*)(w + off); off += 16384 * 2;
    __bf16* p_fc2p = (__bf16*)(w + off); off += 1024  * 2;
    float*  p_b0   = (float*)(w + off);  off += 256 * 4;
    float*  p_b1   = (float*)(w + off);  off += 256 * 4;
    off = al256(off);
    __bf16* p_h1   = (__bf16*)(w + off); off += (size_t)16384 * 1024 * 2;

    // 1) cast/pack weights
    const int prep_total = 802816 + 8192 + 16384 + 16384 + 16384 + 1024 + 256 + 256;
    prep_weights<<<(prep_total + 255) / 256, 256, 0, stream>>>(
        fc1_w, wih0, whh0, wih1, whh1, fc2_w, bih0, bhh0, bih1, bhh1,
        p_fc1w, p_wih0, p_whh0, p_wih1, p_whh1, p_fc2p, p_b0, p_b1);

    // 2) FC1 GEMM: 128 M-blocks x 16 N-strips = 2048 blocks of 8 waves
    fc1_gemm<<<2048, 256, 0, stream>>>(x, fc1_b, p_fc1w, p_h1);

    // 3) fused LSTM x2 + FC2: 128 blocks * 8 waves * 16 rows = 16384
    const size_t smem = (8192 + 16384 + 16384 + 16384 + 1024 + 8 * 2048) * sizeof(__bf16);
    lstm_fused<<<128, 256, smem, stream>>>(p_h1, p_wih0, p_b0, p_b1, fc2_b, out);
}
